// GlobalLocalAttention_47210280517864
// MI455X (gfx1250) — matrix operand order repaired and checked by reading, not compiled, not run on hardware
//
#include <hip/hip_runtime.h>
#include <hip/hip_bf16.h>
#include <math.h>

// ---------------------------------------------------------------------------
// GlobalLocalAttention for MI455X (gfx1250, wave32, WMMA bf16 K=32)
// B=2, C=256, H=W=64 -> N=4096, heads=8, hd=32
//
// Pipeline:
//   prep_weights : w_qkv/w_proj -> bf16 (same layout), w_local -> bf16 [o][tap][c]
//   prep_xT      : x -> bf16 xT[b][n][c]   (LDS-tiled transpose)
//   qkv_gemm     : WMMA GEMM; writes q (scaled), v, and K^T (kT_ws[b][h][n][d])
//   flash_attn   : online-softmax attention, all tiles via async global->LDS DMA
//   conv3_bn     : implicit-GEMM 3x3 conv + BN, tiles via async global->LDS DMA
//   proj_bn      : WMMA GEMM over (attn+local) + BN -> d_out
//
// LDS tile convention (everything gathers as 2x ds_load_b128):
//   A operands: row-major [M][K], K contiguous;  B operands: [N][K], K contiguous
//   rows padded to PAD=40 elems (80B: 16B-aligned chunks, conflict-free banks)
// ---------------------------------------------------------------------------

#define BATCH   2
#define CH      256
#define NSP     4096
#define HEADS   8
#define HD      32
#define EPSV    1e-5f
#define PAD     40

typedef __attribute__((ext_vector_type(16))) __bf16 v16bf;
typedef __attribute__((ext_vector_type(8)))  __bf16 v8bf;
typedef __attribute__((ext_vector_type(8)))  float  v8f;

__device__ __forceinline__ v8f wmma_bf16(v16bf a, v16bf b, v8f c) {
  return __builtin_amdgcn_wmma_f32_16x16x32_bf16(
      false, a, false, b, (short)0, c, false, false);
}

__device__ __forceinline__ __bf16 f2bf(float f) {           // RNE f32->bf16
  unsigned u = __builtin_bit_cast(unsigned, f);
  u += 0x7fffu + ((u >> 16) & 1u);
  return __builtin_bit_cast(__bf16, (unsigned short)(u >> 16));
}
__device__ __forceinline__ unsigned pack_bf2(float a, float b) {
  unsigned ua = __builtin_bit_cast(unsigned, a);
  ua = (ua + 0x7fffu + ((ua >> 16) & 1u)) >> 16;
  unsigned ub = __builtin_bit_cast(unsigned, b);
  ub = (ub + 0x7fffu + ((ub >> 16) & 1u)) & 0xffff0000u;
  return ua | ub;
}

// CDNA5 async global->LDS DMA (16B per lane), tracked by ASYNCcnt
__device__ __forceinline__ void async_b128(void* lds, const void* g) {
  unsigned l = (unsigned)(size_t)lds;        // low 32b of flat = LDS offset
  asm volatile("global_load_async_to_lds_b128 %0, %1, off"
               :: "v"(l), "v"(g) : "memory");
}
__device__ __forceinline__ void wait_async0() {
  asm volatile("s_wait_asynccnt 0x0" ::: "memory");
}

__device__ __forceinline__ v16bf ld_frag2(const __bf16* p0, const __bf16* p1) {
  v8bf lo = *reinterpret_cast<const v8bf*>(p0);
  v8bf hi = *reinterpret_cast<const v8bf*>(p1);
  return __builtin_shufflevector(lo, hi, 0,1,2,3,4,5,6,7,8,9,10,11,12,13,14,15);
}
__device__ __forceinline__ v16bf a_frag(const __bf16* row, int kg) {
  return ld_frag2(row + kg * 16, row + kg * 16 + 8);   // K = kg*16 + i
}
__device__ __forceinline__ v16bf b_frag(const __bf16* rowT, int kg) {
  return ld_frag2(rowT + kg * 16, rowT + kg * 16 + 8); // K = kg*16 + i
}

// ===========================================================================
// Prep 1: weights -> bf16.  wq_bf[768*256], wp_bf[256*256],
//         wfmt[(o*9+tap)*256 + c]  (fragment-ready 3x3 layout)
// ===========================================================================
#define WQ_N (768 * 256)
#define WP_N (256 * 256)
#define WL_N (256 * 9 * 256)
__global__ __launch_bounds__(256)
void prep_weights_kernel(const float* __restrict__ w_qkv,
                         const float* __restrict__ w_local,
                         const float* __restrict__ w_proj,
                         __bf16* __restrict__ wq_bf,
                         __bf16* __restrict__ wp_bf,
                         __bf16* __restrict__ wfmt) {
  size_t t = (size_t)blockIdx.x * 256 + threadIdx.x;
  if (t < WQ_N) {
    wq_bf[t] = f2bf(w_qkv[t]);
  } else if (t < WQ_N + WP_N) {
    size_t u = t - WQ_N;
    wp_bf[u] = f2bf(w_proj[u]);
  } else if (t < WQ_N + WP_N + WL_N) {
    size_t u = t - WQ_N - WP_N;
    int c = (int)(u & 255);
    int row = (int)(u >> 8);            // o*9 + tap
    int o = row / 9, tap = row - o * 9;
    wfmt[u] = f2bf(w_local[(size_t)o * (CH * 9) + c * 9 + tap]);
  }
}

// ===========================================================================
// Prep 2: x (f32 [b][c][n]) -> bf16 xT[b][n][c]  via LDS-tiled transpose
// ===========================================================================
__global__ __launch_bounds__(256)
void prep_xT_kernel(const float* __restrict__ x, __bf16* __restrict__ xT) {
  const int n_base = blockIdx.x * 128;
  const int c_base = blockIdx.y * 32;
  const int b      = blockIdx.z;
  const int tid = threadIdx.x;
  __shared__ __align__(16) __bf16 tl[128 * PAD];
  for (int t = tid; t < 32 * 128; t += 256) {
    int c = t >> 7, n = t & 127;   // coalesced read over n
    tl[n * PAD + c] = f2bf(x[((size_t)b * CH + c_base + c) * NSP + n_base + n]);
  }
  __syncthreads();
  int n = tid & 127, half = (tid >> 7) * 16;
  const uint4* src = reinterpret_cast<const uint4*>(&tl[n * PAD + half]);
  uint4 a0 = src[0], a1 = src[1];
  uint4* dst = reinterpret_cast<uint4*>(
      &xT[((size_t)b * NSP + n_base + n) * CH + c_base + half]);
  dst[0] = a0; dst[1] = a1;
}

// ===========================================================================
// Kernel 1: QKV GEMM [768x256] x [256x4096]; emits q (scaled), K^T, v
// ===========================================================================
__global__ __launch_bounds__(256)
void qkv_gemm_kernel(const __bf16* __restrict__ xT,
                     const __bf16* __restrict__ wq_bf,
                     __bf16* __restrict__ q_ws,
                     __bf16* __restrict__ kT_ws,
                     __bf16* __restrict__ v_ws) {
  const int n_base = blockIdx.x * 128;
  const int o_base = blockIdx.y * 64;
  const int b      = blockIdx.z;
  const int tid  = threadIdx.x;
  const int lane = tid & 31;
  const int wave = tid >> 5;
  const int kg   = lane >> 4;
  const int ln   = lane & 15;
  const int wave_o = wave >> 2;
  const int wave_n = wave & 3;

  __shared__ __align__(16) __bf16 wl[64 * PAD];     // A: [o][c]
  __shared__ __align__(16) __bf16 xlT[128 * PAD];   // B: [n][c]

  v8f acc[2][2];
  acc[0][0] = {}; acc[0][1] = {}; acc[1][0] = {}; acc[1][1] = {};

  for (int c0 = 0; c0 < CH; c0 += 32) {
    __syncthreads();
    {   // A tile: 256 chunks of 16B -> 1 async DMA per thread
      int o_loc = tid >> 2, cs = (tid & 3) * 8;
      async_b128(&wl[o_loc * PAD + cs],
                 wq_bf + (size_t)(o_base + o_loc) * CH + c0 + cs);
    }
#pragma unroll
    for (int j = 0; j < 2; ++j) {   // B tile: 512 chunks -> 2 per thread
      int q = tid + j * 256;
      int n = q >> 2, cs = (q & 3) * 8;
      async_b128(&xlT[n * PAD + cs],
                 xT + ((size_t)b * NSP + n_base + n) * CH + c0 + cs);
    }
    wait_async0();
    __syncthreads();

    v16bf afr[2], bfr[2];
#pragma unroll
    for (int ti = 0; ti < 2; ++ti)
      afr[ti] = a_frag(&wl[(wave_o * 32 + ti * 16 + ln) * PAD], kg);
#pragma unroll
    for (int tj = 0; tj < 2; ++tj)
      bfr[tj] = b_frag(&xlT[(wave_n * 32 + tj * 16 + ln) * PAD], kg);
#pragma unroll
    for (int ti = 0; ti < 2; ++ti)
#pragma unroll
      for (int tj = 0; tj < 2; ++tj)
        acc[ti][tj] = wmma_bf16(afr[ti], bfr[tj], acc[ti][tj]);
  }

  const float scale = 0.176776695296636881f;  // 32^-0.5
#pragma unroll
  for (int ti = 0; ti < 2; ++ti)
#pragma unroll
    for (int tj = 0; tj < 2; ++tj)
#pragma unroll
      for (int r = 0; r < 8; ++r) {
        int o = o_base + wave_o * 32 + ti * 16 + r + 8 * kg;
        int n = n_base + wave_n * 32 + tj * 16 + ln;
        int qi = o >> 8;      // 0=q 1=k 2=v
        int ch = o & 255;
        float v = acc[ti][tj][r];
        if (qi == 0) {
          q_ws[((size_t)b * CH + ch) * NSP + n] = f2bf(v * scale);
        } else if (qi == 1) {
          int h = ch >> 5, d = ch & 31;      // K stored transposed: [b][h][n][d]
          kT_ws[(((size_t)b * HEADS + h) * NSP + n) * HD + d] = f2bf(v);
        } else {
          v_ws[((size_t)b * CH + ch) * NSP + n] = f2bf(v);
        }
      }
}

// ===========================================================================
// Kernel 2: flash attention per (b,h).  All tile staging is async DMA.
// ===========================================================================
__global__ __launch_bounds__(128)
void flash_attn_kernel(const __bf16* __restrict__ q_ws,
                       const __bf16* __restrict__ kT_ws,
                       const __bf16* __restrict__ v_ws,
                       float* __restrict__ attn_out) {
  const int nblk = blockIdx.x;
  const int h    = blockIdx.y;
  const int b    = blockIdx.z;
  const int tid  = threadIdx.x;
  const int lane = tid & 31;
  const int wave = tid >> 5;
  const int kg   = lane >> 4;
  const int ln   = lane & 15;

  const __bf16* qp  = q_ws  + ((size_t)b * CH + h * HD) * NSP;
  const __bf16* kTp = kT_ws + ((size_t)b * HEADS + h) * NSP * HD;
  const __bf16* vp  = v_ws  + ((size_t)b * CH + h * HD) * NSP;

  __shared__ __align__(16) __bf16 kT[32 * PAD];     // A (S^T): [m][d]
  __shared__ __align__(16) __bf16 vlds[32 * PAD];   // A (O):   [d][m]
  __shared__ __align__(16) __bf16 pT[4][16 * PAD];  // B (O) per wave: [n][m]

  const int n0 = nblk * 64 + wave * 16;

  v16bf bq;   // persistent B operand of Q: column n, K = d
#pragma unroll
  for (int i = 0; i < 16; ++i) bq[i] = qp[(size_t)(kg * 16 + i) * NSP + n0 + ln];

  v8f oacc[2]; oacc[0] = {}; oacc[1] = {};
  float m_run = -1e30f, l_run = 0.0f;

  const int sr = tid >> 2;        // staging row 0..31
  const int sg = tid & 3;         // 16B segment

  for (int m0 = 0; m0 < NSP; m0 += 32) {
    __syncthreads();
    // K^T tile [m][d]: contiguous rows of kT_ws -> async
    async_b128(&kT[sr * PAD + sg * 8], kTp + (size_t)(m0 + sr) * HD + sg * 8);
    // V tile [d][m]: contiguous in m -> async
    async_b128(&vlds[sr * PAD + sg * 8], vp + (size_t)sr * NSP + m0 + sg * 8);
    wait_async0();
    __syncthreads();

    v8f st[2];
#pragma unroll
    for (int t = 0; t < 2; ++t) {
      v16bf ak = a_frag(&kT[(t * 16 + ln) * PAD], kg);
      v8f z = {};
      st[t] = wmma_bf16(ak, bq, z);
    }

    float lm = -1e30f;
#pragma unroll
    for (int t = 0; t < 2; ++t)
#pragma unroll
      for (int r = 0; r < 8; ++r) lm = fmaxf(lm, st[t][r]);
    lm = fmaxf(lm, __shfl_xor(lm, 16, 32));
    float m_new = fmaxf(m_run, lm);
    float corr  = __expf(m_run - m_new);
    float psum = 0.0f;
#pragma unroll
    for (int t = 0; t < 2; ++t)
#pragma unroll
      for (int r = 0; r < 8; ++r) {
        float p = __expf(st[t][r] - m_new);
        psum += p;
        int m = t * 16 + r + 8 * kg;
        pT[wave][ln * PAD + m] = f2bf(p);
      }
    psum += __shfl_xor(psum, 16, 32);
    l_run = l_run * corr + psum;
    m_run = m_new;
#pragma unroll
    for (int t = 0; t < 2; ++t)
#pragma unroll
      for (int r = 0; r < 8; ++r) oacc[t][r] *= corr;

    asm volatile("s_wait_dscnt 0x0" ::: "memory");  // wave-sync P reuse

    v16bf bp = b_frag(&pT[wave][ln * PAD], kg);
#pragma unroll
    for (int t = 0; t < 2; ++t) {
      v16bf av = a_frag(&vlds[(t * 16 + ln) * PAD], kg);
      oacc[t] = wmma_bf16(av, bp, oacc[t]);
    }
  }

  float inv_l = 1.0f / l_run;
#pragma unroll
  for (int t = 0; t < 2; ++t)
#pragma unroll
    for (int r = 0; r < 8; ++r) {
      int dd = t * 16 + r + 8 * kg;
      attn_out[((size_t)b * CH + h * HD + dd) * NSP + n0 + ln] =
          oacc[t][r] * inv_l;
    }
}

// ===========================================================================
// Kernel 3: 3x3 conv (implicit GEMM) + BN.  Both tiles staged via async DMA.
// ===========================================================================
__global__ __launch_bounds__(256)
void conv3_bn_kernel(const __bf16* __restrict__ xT,
                     const __bf16* __restrict__ wfmt,
                     const float* __restrict__ gamma, const float* __restrict__ beta,
                     const float* __restrict__ mean,  const float* __restrict__ var,
                     float* __restrict__ out) {
  const int y      = blockIdx.x;
  const int o_base = blockIdx.y * 64;
  const int b      = blockIdx.z;
  const int tid  = threadIdx.x;
  const int lane = tid & 31;
  const int wave = tid >> 5;
  const int kg   = lane >> 4;
  const int ln   = lane & 15;
  const int wave_o = wave >> 1;
  const int xsub0  = (wave & 1) * 2;

  __shared__ __align__(16) __bf16 wl[576 * PAD];      // A: [o*9+tap][c]
  __shared__ __align__(16) __bf16 xlT[198 * PAD];     // B: [xi*3+ky][c]

  v8f acc[2]; acc[0] = {}; acc[1] = {};

  for (int c0 = 0; c0 < CH; c0 += 32) {
    if (c0 + 32 < CH)
      __builtin_prefetch(wfmt + ((size_t)o_base * 9 + (tid & 255)) * CH
                              + (c0 + 32), 0, 1);
    __syncthreads();
    // W tile: 576 rows x 4 chunks = 9 async per thread
#pragma unroll
    for (int j = 0; j < 9; ++j) {
      int q = tid + j * 256;
      int row = q >> 2, cs = (q & 3) * 8;
      async_b128(&wl[row * PAD + cs],
                 wfmt + ((size_t)o_base * 9 + row) * CH + c0 + cs);
    }
    // X tile: 198 rows x 4 chunks; async for in-bounds, zero-fill else
    for (int q = tid; q < 198 * 4; q += 256) {
      int row = q >> 2, cs = (q & 3) * 8;
      int xi = row / 3, ky = row - xi * 3;
      int yy = y + ky - 1, xx = xi - 1;
      if (yy >= 0 && yy < 64 && xx >= 0 && xx < 64) {
        async_b128(&xlT[row * PAD + cs],
                   xT + ((size_t)b * NSP + yy * 64 + xx) * CH + c0 + cs);
      } else {
        uint4 z = make_uint4(0, 0, 0, 0);
        *reinterpret_cast<uint4*>(&xlT[row * PAD + cs]) = z;
      }
    }
    wait_async0();
    asm volatile("s_wait_dscnt 0x0" ::: "memory");
    __syncthreads();

#pragma unroll
    for (int tap = 0; tap < 9; ++tap) {
      int ky = tap / 3, kx = tap - ky * 3;
      v16bf afr = a_frag(&wl[((wave_o * 16 + ln) * 9 + tap) * PAD], kg);
#pragma unroll
      for (int s = 0; s < 2; ++s) {
        int col = (xsub0 + s) * 16 + ln + kx;   // xi index (x = col-1)
        v16bf bfr = b_frag(&xlT[(col * 3 + ky) * PAD], kg);
        acc[s] = wmma_bf16(afr, bfr, acc[s]);
      }
    }
  }

#pragma unroll
  for (int s = 0; s < 2; ++s)
#pragma unroll
    for (int r = 0; r < 8; ++r) {
      int o  = o_base + wave_o * 16 + r + 8 * kg;
      int xp = (xsub0 + s) * 16 + ln;
      float inv = gamma[o] * rsqrtf(var[o] + EPSV);
      float val = acc[s][r] * inv + (beta[o] - mean[o] * inv);
      out[(((size_t)b * CH + o) * 64 + y) * 64 + xp] = val;
    }
}

// ===========================================================================
// Kernel 4: proj GEMM over (attn + local) + BN -> d_out (f32)
// ===========================================================================
__global__ __launch_bounds__(256)
void proj_bn_kernel(const float* __restrict__ attn_in,
                    const float* __restrict__ local_in,
                    const __bf16* __restrict__ wp_bf,
                    const float* __restrict__ gamma, const float* __restrict__ beta,
                    const float* __restrict__ mean,  const float* __restrict__ var,
                    float* __restrict__ out) {
  const int n_base = blockIdx.x * 128;
  const int o_base = blockIdx.y * 64;
  const int b      = blockIdx.z;
  const int tid  = threadIdx.x;
  const int lane = tid & 31;
  const int wave = tid >> 5;
  const int kg   = lane >> 4;
  const int ln   = lane & 15;
  const int wave_o = wave >> 2;
  const int wave_n = wave & 3;

  __shared__ __align__(16) __bf16 wl[64 * PAD];
  __shared__ __align__(16) __bf16 xlT[128 * PAD];

  v8f acc[2][2];
  acc[0][0] = {}; acc[0][1] = {}; acc[1][0] = {}; acc[1][1] = {};

  for (int c0 = 0; c0 < CH; c0 += 32) {
    __syncthreads();
    {   // A tile via async DMA
      int o_loc = tid >> 2, cs = (tid & 3) * 8;
      async_b128(&wl[o_loc * PAD + cs],
                 wp_bf + (size_t)(o_base + o_loc) * CH + c0 + cs);
    }
    {   // B tile: sum two f32 tensors, pack pairs, 2x ds_store_b128
      int n = tid & 127, ch0 = (tid >> 7) * 16;
      unsigned u[8];
#pragma unroll
      for (int j = 0; j < 8; ++j) {
        size_t i0 = ((size_t)b * CH + c0 + ch0 + 2 * j) * NSP + n_base + n;
        float a0 = attn_in[i0] + local_in[i0];
        float a1 = attn_in[i0 + NSP] + local_in[i0 + NSP];
        u[j] = pack_bf2(a0, a1);
      }
      uint4* dst = reinterpret_cast<uint4*>(&xlT[n * PAD + ch0]);
      dst[0] = make_uint4(u[0], u[1], u[2], u[3]);
      dst[1] = make_uint4(u[4], u[5], u[6], u[7]);
    }
    wait_async0();
    __syncthreads();

    v16bf afr[2], bfr[2];
#pragma unroll
    for (int ti = 0; ti < 2; ++ti)
      afr[ti] = a_frag(&wl[(wave_o * 32 + ti * 16 + ln) * PAD], kg);
#pragma unroll
    for (int tj = 0; tj < 2; ++tj)
      bfr[tj] = b_frag(&xlT[(wave_n * 32 + tj * 16 + ln) * PAD], kg);
#pragma unroll
    for (int ti = 0; ti < 2; ++ti)
#pragma unroll
      for (int tj = 0; tj < 2; ++tj)
        acc[ti][tj] = wmma_bf16(afr[ti], bfr[tj], acc[ti][tj]);
  }

#pragma unroll
  for (int ti = 0; ti < 2; ++ti)
#pragma unroll
    for (int tj = 0; tj < 2; ++tj)
#pragma unroll
      for (int r = 0; r < 8; ++r) {
        int o = o_base + wave_o * 32 + ti * 16 + r + 8 * kg;
        int n = n_base + wave_n * 32 + tj * 16 + ln;
        float inv = gamma[o] * rsqrtf(var[o] + EPSV);
        float val = acc[ti][tj][r] * inv + (beta[o] - mean[o] * inv);
        out[((size_t)b * CH + o) * NSP + n] = val;
      }
}

// ===========================================================================
// Launch
// ===========================================================================
extern "C" void kernel_launch(void* const* d_in, const int* in_sizes, int n_in,
                              void* d_out, int out_size, void* d_ws, size_t ws_size,
                              hipStream_t stream) {
  const float* x           = (const float*)d_in[0];
  const float* w_qkv       = (const float*)d_in[1];
  const float* w_local     = (const float*)d_in[2];
  const float* local_gamma = (const float*)d_in[3];
  const float* local_beta  = (const float*)d_in[4];
  const float* local_mean  = (const float*)d_in[5];
  const float* local_var   = (const float*)d_in[6];
  const float* w_proj      = (const float*)d_in[7];
  const float* proj_gamma  = (const float*)d_in[8];
  const float* proj_beta   = (const float*)d_in[9];
  const float* proj_mean   = (const float*)d_in[10];
  const float* proj_var    = (const float*)d_in[11];
  float* out = (float*)d_out;

  // ---- workspace partition (~35 MB) ----
  char* p = (char*)d_ws;
  auto take = [&](size_t bytes) {
    char* r = p; p += (bytes + 255) & ~(size_t)255; return r;
  };
  __bf16* wq_bf   = (__bf16*)take((size_t)WQ_N * 2);
  __bf16* wp_bf   = (__bf16*)take((size_t)WP_N * 2);
  __bf16* wfmt    = (__bf16*)take((size_t)WL_N * 2);
  __bf16* xT      = (__bf16*)take((size_t)BATCH * NSP * CH * 2);
  __bf16* q_ws    = (__bf16*)take((size_t)BATCH * CH * NSP * 2);
  __bf16* kT_ws   = (__bf16*)take((size_t)BATCH * CH * NSP * 2);
  __bf16* v_ws    = (__bf16*)take((size_t)BATCH * CH * NSP * 2);
  float*  attn_ws = (float*) take((size_t)BATCH * CH * NSP * 4);
  float*  local_ws= (float*) take((size_t)BATCH * CH * NSP * 4);

  const int prep_total = WQ_N + WP_N + WL_N;
  prep_weights_kernel<<<(prep_total + 255) / 256, 256, 0, stream>>>(
      w_qkv, w_local, w_proj, wq_bf, wp_bf, wfmt);

  prep_xT_kernel<<<dim3(NSP / 128, CH / 32, BATCH), 256, 0, stream>>>(x, xT);

  qkv_gemm_kernel<<<dim3(NSP / 128, (3 * CH) / 64, BATCH), 256, 0, stream>>>(
      xT, wq_bf, q_ws, kT_ws, v_ws);

  flash_attn_kernel<<<dim3(NSP / 64, HEADS, BATCH), 128, 0, stream>>>(
      q_ws, kT_ws, v_ws, attn_ws);

  conv3_bn_kernel<<<dim3(64, CH / 64, BATCH), 256, 0, stream>>>(
      xT, wfmt, local_gamma, local_beta, local_mean, local_var, local_ws);

  proj_bn_kernel<<<dim3(NSP / 128, CH / 64, BATCH), 256, 0, stream>>>(
      attn_ws, local_ws, wp_bf, proj_gamma, proj_beta, proj_mean, proj_var, out);
}
